// ChunkedAttention_11622181503352
// MI455X (gfx1250) — compile-verified, matrix operand order checked by reading, not run on hardware
//
#include <hip/hip_runtime.h>
#include <hip/hip_bf16.h>

#define DIM 256
#define CHUNK 128
#define OVERLAP 32
#define CSTRIDE 96
#define HEADS 8
#define HEAD_DIM 32
#define BATCH 4
#define SEQ 16384
#define NCHUNK 171                      // ceil((16384-32)/96)
#define TPAD (NCHUNK*CSTRIDE + OVERLAP) // 16448

typedef __attribute__((ext_vector_type(16))) __bf16 v16bf;
typedef __attribute__((ext_vector_type(8)))  float  v8f;
typedef unsigned short u16;
typedef unsigned int   u32;

__device__ __forceinline__ u16 f2bf(float f) {
  u32 u = __builtin_bit_cast(u32, f);
  u32 r = (u + 0x7FFFu + ((u >> 16) & 1u)) >> 16;   // round-to-nearest-even
  return (u16)r;
}

__device__ __forceinline__ u32 pack2bf(float lo, float hi) {
  return (u32)f2bf(lo) | ((u32)f2bf(hi) << 16);
}

union FragBF { u32 u[8]; v16bf v; };

// 16x32 bf16 fragment; element (i,k) at base[i*ld + k].
// ISA 16-bit A layout: lanes 0-15 hold row i=lane, K=0..7 & 16..23;
// lanes 16-31 hold K=8..15 & 24..31. Also used for B tiles stored
// K-contiguous per output column (B(k,n) = base[n*ld + k]).
__device__ __forceinline__ v16bf load_frag_kcontig(const u16* __restrict__ base, int ld) {
  const int lane = threadIdx.x & 31;
  const u16* row = base + (lane & 15) * ld;
  const int k0 = (lane >> 4) * 8;
  FragBF f;
#pragma unroll
  for (int i = 0; i < 4; ++i) {
    int k = k0 + 2 * i;
    f.u[i]     = (u32)row[k]      | ((u32)row[k + 1]  << 16);
    f.u[i + 4] = (u32)row[k + 16] | ((u32)row[k + 17] << 16);
  }
  return f.v;
}

__device__ __forceinline__ v8f wmma_bf16(v16bf a, v16bf b, v8f c) {
  return __builtin_amdgcn_wmma_f32_16x16x32_bf16(false, a, false, b, (short)0, c, false, false);
}

__global__ void cvt_weights(const float* __restrict__ w_in, const float* __restrict__ w_out,
                            u16* __restrict__ w_in_bf, u16* __restrict__ w_out_bf) {
  int i = blockIdx.x * blockDim.x + threadIdx.x;
  if (i < 3 * DIM * DIM) w_in_bf[i] = f2bf(w_in[i]);
  if (i < DIM * DIM)     w_out_bf[i] = f2bf(w_out[i]);
}

__launch_bounds__(256, 1)
__global__ void chunk_attn_kernel(const float* __restrict__ x,
                                  const u16*  __restrict__ w_in,   // (768,256) bf16
                                  const float* __restrict__ b_in,  // (768)
                                  const u16*  __restrict__ w_out,  // (256,256) bf16
                                  const float* __restrict__ b_out, // (256)
                                  float* __restrict__ recon) {
  extern __shared__ u16 smem[];
  u16* Xb = smem;                  // 128x256 bf16 chunk input; reused for O
  u16* Qb = Xb + CHUNK * DIM;      // 128x256 token-major
  u16* Kb = Qb + CHUNK * DIM;      // 128x256 token-major
  u16* Vt = Kb + CHUNK * DIM;      // 256x128 feature-major (transposed!)
  u16* Pb = Vt + CHUNK * DIM;      // 8 waves x (16x128) softmax tiles

  const int b    = blockIdx.x / NCHUNK;
  const int l    = blockIdx.x % NCHUNK;
  const int tid  = threadIdx.x;
  const int wave = tid >> 5;
  const int lane = tid & 31;
  const int col  = lane & 15;
  const int r0   = (lane >> 4) * 8;
  const int m0   = wave * 16;      // each wave owns one 16-row M block

  // ---- Stage 0: gather chunk, f32 -> bf16 into LDS (zero-pad past SEQ) ----
  for (int i = tid; i < CHUNK * DIM / 4; i += 256) {
    int r = i >> 6;                // DIM/4 = 64 float4 per row
    int c = (i & 63) * 4;
    int t = l * CSTRIDE + r;
    float4 v = make_float4(0.f, 0.f, 0.f, 0.f);
    if (t < SEQ) v = *(const float4*)&x[((size_t)b * SEQ + t) * DIM + c];
    *(uint2*)&Xb[r * DIM + c] = make_uint2(pack2bf(v.x, v.y), pack2bf(v.z, v.w));
  }
  __syncthreads();

  // ---- Stage 1: QKV = X * W_in^T + b_in ----
  // One loop over the three Q/K/V segments (no divergent dst select).
#pragma unroll 1
  for (int seg = 0; seg < 3; ++seg) {
#pragma unroll 1
    for (int nt = 0; nt < 16; ++nt) {
      const int e0 = (seg * 16 + nt) * 16;
      const float bias = b_in[e0 + col];
      v8f acc;
#pragma unroll
      for (int r = 0; r < 8; ++r) acc[r] = bias;
#pragma unroll
      for (int kb = 0; kb < DIM / 32; ++kb) {
        v16bf a = load_frag_kcontig(&Xb[m0 * DIM + kb * 32], DIM);
        v16bf w = load_frag_kcontig(&w_in[(size_t)e0 * DIM + kb * 32], DIM);
        acc = wmma_bf16(a, w, acc);
      }
      if (seg < 2) {
        u16* dst = (seg == 0) ? &Qb[m0 * DIM + e0] : &Kb[m0 * DIM + (e0 - DIM)];
#pragma unroll
        for (int r = 0; r < 8; ++r) dst[(r0 + r) * DIM + col] = f2bf(acc[r]);
      } else {
        // Transposed store: Vt[feat][token]; 8 consecutive tokens -> b128 store
        const int feat = (e0 - 2 * DIM) + col;
        uint4 p;
        p.x = pack2bf(acc[0], acc[1]);
        p.y = pack2bf(acc[2], acc[3]);
        p.z = pack2bf(acc[4], acc[5]);
        p.w = pack2bf(acc[6], acc[7]);
        *(uint4*)&Vt[feat * CHUNK + m0 + r0] = p;
      }
    }
  }
  __syncthreads();

  // ---- Stage 2: per-head attention (wave == head) ----
  const int h = wave;
  u16* P = &Pb[wave * 16 * CHUNK];
  u16* Ob = Xb;                                   // X no longer needed
  const float scale = 0.17677669529663687f;       // 1/sqrt(HEAD_DIM)

#pragma unroll 1
  for (int mt = 0; mt < 8; ++mt) {
    const int q0 = mt * 16;
    v16bf qa = load_frag_kcontig(&Qb[q0 * DIM + h * HEAD_DIM], DIM);
    float s[8][8];
#pragma unroll
    for (int nt = 0; nt < 8; ++nt) {
      v16bf kf = load_frag_kcontig(&Kb[nt * 16 * DIM + h * HEAD_DIM], DIM);
      v8f z;
#pragma unroll
      for (int r = 0; r < 8; ++r) z[r] = 0.f;
      v8f acc = wmma_bf16(qa, kf, z);
#pragma unroll
      for (int r = 0; r < 8; ++r) s[nt][r] = acc[r] * scale;
    }
    // Row softmax: reduce over 8 N-tiles per lane, then across the 16 lanes
    // of each wave32 half (rows differ between halves -> xor offsets < 16).
#pragma unroll
    for (int r = 0; r < 8; ++r) {
      float m = s[0][r];
#pragma unroll
      for (int nt = 1; nt < 8; ++nt) m = fmaxf(m, s[nt][r]);
#pragma unroll
      for (int off = 1; off < 16; off <<= 1) m = fmaxf(m, __shfl_xor(m, off, 32));
      float t = 0.f;
#pragma unroll
      for (int nt = 0; nt < 8; ++nt) { s[nt][r] = __expf(s[nt][r] - m); t += s[nt][r]; }
#pragma unroll
      for (int off = 1; off < 16; off <<= 1) t += __shfl_xor(t, off, 32);
      float inv = 1.f / t;
#pragma unroll
      for (int nt = 0; nt < 8; ++nt) s[nt][r] *= inv;
    }
#pragma unroll
    for (int nt = 0; nt < 8; ++nt)
#pragma unroll
      for (int r = 0; r < 8; ++r)
        P[(r0 + r) * CHUNK + nt * 16 + col] = f2bf(s[nt][r]);

    // O tile (16x32) = P (16x128) x V_h (128x32); V is feature-major so the
    // B fragment is K-contiguous and loads as ds_load_b128.
#pragma unroll
    for (int nt2 = 0; nt2 < 2; ++nt2) {
      v8f acc;
#pragma unroll
      for (int r = 0; r < 8; ++r) acc[r] = 0.f;
#pragma unroll
      for (int kb = 0; kb < 4; ++kb) {
        v16bf pa = load_frag_kcontig(&P[kb * 32], CHUNK);
        v16bf vb = load_frag_kcontig(&Vt[(h * HEAD_DIM + nt2 * 16) * CHUNK + kb * 32], CHUNK);
        acc = wmma_bf16(pa, vb, acc);
      }
#pragma unroll
      for (int r = 0; r < 8; ++r)
        Ob[(q0 + r0 + r) * DIM + h * HEAD_DIM + nt2 * 16 + col] = f2bf(acc[r]);
    }
  }
  __syncthreads();

  // ---- Stage 3: out projection + overlap-add via float atomics ----
#pragma unroll 1
  for (int nt = 0; nt < 16; ++nt) {
    const int e0 = nt * 16;
    const float bias = b_out[e0 + col];
    v8f acc;
#pragma unroll
    for (int r = 0; r < 8; ++r) acc[r] = bias;
#pragma unroll
    for (int kb = 0; kb < DIM / 32; ++kb) {
      v16bf a = load_frag_kcontig(&Ob[m0 * DIM + kb * 32], DIM);
      v16bf w = load_frag_kcontig(&w_out[(size_t)e0 * DIM + kb * 32], DIM);
      acc = wmma_bf16(a, w, acc);
    }
    const int e = e0 + col;
#pragma unroll
    for (int r = 0; r < 8; ++r) {
      int t = l * CSTRIDE + m0 + r0 + r;   // always < TPAD
      atomicAdd(&recon[((size_t)b * TPAD + t) * DIM + e], acc[r]);
    }
  }
}

__global__ void finalize_kernel(const float* __restrict__ recon, float* __restrict__ out) {
  size_t i = (size_t)blockIdx.x * blockDim.x + threadIdx.x;
  const size_t total = (size_t)BATCH * SEQ * DIM;
  if (i >= total) return;
  int d = (int)(i % DIM);
  int t = (int)((i / DIM) % SEQ);
  int b = (int)(i / ((size_t)SEQ * DIM));
  int lmax = t / CSTRIDE; if (lmax > NCHUNK - 1) lmax = NCHUNK - 1;
  int lmin = (t >= CHUNK) ? ((t - CHUNK) / CSTRIDE + 1) : 0;
  float cnt = (float)(lmax - lmin + 1);
  out[i] = recon[((size_t)b * TPAD + t) * DIM + d] / (cnt + 1e-6f);
}

extern "C" void kernel_launch(void* const* d_in, const int* in_sizes, int n_in,
                              void* d_out, int out_size, void* d_ws, size_t ws_size,
                              hipStream_t stream) {
  (void)in_sizes; (void)n_in; (void)out_size; (void)ws_size;
  const float* x     = (const float*)d_in[0];
  const float* w_in  = (const float*)d_in[1];
  const float* b_in  = (const float*)d_in[2];
  const float* w_out = (const float*)d_in[3];
  const float* b_out = (const float*)d_in[4];
  float* out = (float*)d_out;

  float* recon = (float*)d_ws;                       // B*TPAD*DIM f32 accumulator
  const size_t reconElems = (size_t)BATCH * TPAD * DIM;
  u16* w_in_bf  = (u16*)(recon + reconElems);        // 768x256 bf16
  u16* w_out_bf = w_in_bf + 3 * DIM * DIM;           // 256x256 bf16

  hipMemsetAsync(recon, 0, reconElems * sizeof(float), stream);
  cvt_weights<<<(3 * DIM * DIM + 255) / 256, 256, 0, stream>>>(w_in, w_out, w_in_bf, w_out_bf);

  const int smem_bytes = (4 * CHUNK * DIM + HEADS * 16 * CHUNK) * (int)sizeof(u16); // 288 KB
  hipFuncSetAttribute((const void*)chunk_attn_kernel,
                      hipFuncAttributeMaxDynamicSharedMemorySize, smem_bytes);
  chunk_attn_kernel<<<BATCH * NCHUNK, 256, smem_bytes, stream>>>(
      x, w_in_bf, b_in, w_out_bf, b_out, recon);

  const size_t total = (size_t)BATCH * SEQ * DIM;
  finalize_kernel<<<(int)((total + 255) / 256), 256, 0, stream>>>(recon, out);
}
